// NeuralMemory_79293686219158
// MI455X (gfx1250) — matrix operand order repaired and checked by reading, not compiled, not run on hardware
//
#include <hip/hip_runtime.h>
#include <math.h>

#define Bsz 4
#define Ssz 4096
#define Dsz 512
#define CH  64
#define NCH 64

typedef __attribute__((ext_vector_type(16))) __bf16 v16bf;
typedef __attribute__((ext_vector_type(8)))  float  v8f;

static_assert(sizeof(__bf16) == 2, "bf16 size");

__device__ inline v8f v8zero() {
  v8f z;
#pragma unroll
  for (int i = 0; i < 8; ++i) z[i] = 0.0f;
  return z;
}

__device__ inline v8f wmma_bf16(v16bf a, v16bf b, v8f c) {
  // D = A(16x32 bf16) * B(32x16 bf16) + C(16x16 f32)
  return __builtin_amdgcn_wmma_f32_16x16x32_bf16(false, a, false, b, (short)0, c,
                                                 false, false);
}

// A-fragment 16x32 bf16 from LDS tile (row-major, `stride` bf16 per row).
// lane<16: row=lane, K={0..7,16..23}; lane>=16: row=lane-16, K={8..15,24..31}
__device__ inline v16bf ld_frag_a(const __bf16* tile, int stride, int lane) {
  const int m  = lane & 15;
  const int kh = (lane >> 4) << 3;
  const __bf16* p = tile + m * stride + kh;
  union { v16bf v; uint4 q[2]; } u;
  u.q[0] = *(const uint4*)p;
  u.q[1] = *(const uint4*)(p + 16);
  return u.v;
}

// B-fragment 32x16 bf16 from LDS tile (row-major): col=lane&15, K={0..7,16..23} / {8..15,24..31}
__device__ inline v16bf ld_frag_b(const __bf16* tile, int stride, int lane) {
  const int n  = lane & 15;
  const int kh = (lane >> 4) << 3;
  v16bf v;
#pragma unroll
  for (int i = 0; i < 8; ++i) {
    v[i]     = tile[(kh + i) * stride + n];
    v[8 + i] = tile[(16 + kh + i) * stride + n];
  }
  return v;
}

// ---------------------------------------------------------------------------
// Projection GEMM: P = x @ W  ([16384,512] x [512,512]), z selects Wk/Wv/Wq,
// outputs bf16. Per-WG tile 128x64, 8 waves, each wave 32x32 (2x2 WMMA tiles).
// ---------------------------------------------------------------------------
__global__ __launch_bounds__(256) void proj_kernel(
    const float* __restrict__ x, const float* __restrict__ Wk,
    const float* __restrict__ Wv, const float* __restrict__ Wq,
    __bf16* __restrict__ kb, __bf16* __restrict__ vb, __bf16* __restrict__ qb) {
  __shared__ __bf16 As[128][40];  // 128x32 x-tile (+pad)
  __shared__ __bf16 Bs[32][72];   // 32x64 W-tile (+pad)

  const int tid = threadIdx.x, lane = tid & 31, w = tid >> 5;
  const int rb = blockIdx.x * 128;
  const int cb = blockIdx.y * 64;
  const float* W;
  __bf16* dst;
  if (blockIdx.z == 0)      { W = Wk; dst = kb; }
  else if (blockIdx.z == 1) { W = Wv; dst = vb; }
  else                      { W = Wq; dst = qb; }

  const int wr = w & 3, wc = w >> 2;
  v8f acc[2][2];
#pragma unroll
  for (int i = 0; i < 2; ++i)
#pragma unroll
    for (int j = 0; j < 2; ++j) acc[i][j] = v8zero();

  const int arow = tid >> 1, ahalf = (tid & 1) * 16;
  const int brow = tid >> 3, bseg = (tid & 7) * 8;

  for (int kk = 0; kk < Dsz; kk += 32) {
    // stage A (f32 -> bf16): 16 floats / thread
    {
      const float* src = x + (size_t)(rb + arow) * Dsz + kk + ahalf;
      union { __bf16 h[16]; uint4 q[2]; } u;
#pragma unroll
      for (int i = 0; i < 4; ++i) {
        float4 f = *(const float4*)(src + i * 4);
        u.h[i * 4 + 0] = (__bf16)f.x; u.h[i * 4 + 1] = (__bf16)f.y;
        u.h[i * 4 + 2] = (__bf16)f.z; u.h[i * 4 + 3] = (__bf16)f.w;
      }
      *(uint4*)&As[arow][ahalf]     = u.q[0];
      *(uint4*)&As[arow][ahalf + 8] = u.q[1];
    }
    // stage B: 8 floats / thread
    {
      const float* src = W + (size_t)(kk + brow) * Dsz + cb + bseg;
      union { __bf16 h[8]; uint4 q; } u;
      float4 f0 = *(const float4*)src, f1 = *(const float4*)(src + 4);
      u.h[0] = (__bf16)f0.x; u.h[1] = (__bf16)f0.y; u.h[2] = (__bf16)f0.z;
      u.h[3] = (__bf16)f0.w; u.h[4] = (__bf16)f1.x; u.h[5] = (__bf16)f1.y;
      u.h[6] = (__bf16)f1.z; u.h[7] = (__bf16)f1.w;
      *(uint4*)&Bs[brow][bseg] = u.q;
    }
    __syncthreads();
    v16bf a0 = ld_frag_a(&As[wr * 32][0], 40, lane);
    v16bf a1 = ld_frag_a(&As[wr * 32 + 16][0], 40, lane);
    v16bf b0 = ld_frag_b(&Bs[0][wc * 32], 72, lane);
    v16bf b1 = ld_frag_b(&Bs[0][wc * 32 + 16], 72, lane);
    acc[0][0] = wmma_bf16(a0, b0, acc[0][0]);
    acc[0][1] = wmma_bf16(a0, b1, acc[0][1]);
    acc[1][0] = wmma_bf16(a1, b0, acc[1][0]);
    acc[1][1] = wmma_bf16(a1, b1, acc[1][1]);
    __syncthreads();
  }

  const int kh = (lane >> 4) << 3, nn = lane & 15;
#pragma unroll
  for (int mt = 0; mt < 2; ++mt)
#pragma unroll
    for (int nt = 0; nt < 2; ++nt)
#pragma unroll
      for (int r = 0; r < 8; ++r) {
        int row = rb + wr * 32 + mt * 16 + kh + r;
        int col = cb + wc * 32 + nt * 16 + nn;
        dst[(size_t)row * Dsz + col] = (__bf16)acc[mt][nt][r];
      }
}

// ---------------------------------------------------------------------------
// Recurrent step kernel: one WG per (batch, 64-col block of M). Runs all 64
// chunk steps internally — the update is column-separable, so no inter-WG sync.
// M/Smom live in VGPRs (wave w owns rows [32w,32w+32) of its column block).
// ---------------------------------------------------------------------------
#define MB_ST 72
#define SM_AK 73728                    // 512*72*2
#define SM_AQ (SM_AK + 5120)           // 64*40*2
#define SM_GR SM_AK                    // grad overlays ak/aq after phase B
#define SM_RED (SM_AK + 10240)
#define SMEM_BYTES (SM_RED + 64)

__global__ __launch_bounds__(512) void step_kernel(
    const __bf16* __restrict__ kb, const __bf16* __restrict__ vb,
    const __bf16* __restrict__ qb, float* __restrict__ out,
    float* __restrict__ surp) {
  extern __shared__ char smraw[];
  __bf16* mb = (__bf16*)smraw;            // M block bf16 [512][72]; reused as k^T
  __bf16* gr = (__bf16*)(smraw + SM_GR);  // grad' [64][72] (overlays ak/aq)
  float*  red = (float*)(smraw + SM_RED);

  const int tid = threadIdx.x, lane = tid & 31, w = tid >> 5;
  const int J = blockIdx.x * 64;
  const int b = blockIdx.y;
  const int kh = (lane >> 4) << 3;
  const int nn = lane & 15;

  const bool isPred = (w < 8);
  const int ww = w & 7;
  const int strip = (ww & 3) * 16;  // chunk-row strip for phase B
  const int nhalf = ww >> 2;        // which 32-col half

  v8f Mreg[8], Sreg[8];
#pragma unroll
  for (int f = 0; f < 8; ++f) { Mreg[f] = v8zero(); Sreg[f] = v8zero(); }

#pragma unroll 1
  for (int t = 0; t < NCH; ++t) {
    const int crow = t * CH;
    // ---- phase A: publish M (bf16) into LDS for use as WMMA B operand ----
    __syncthreads();  // previous step's k^T readers done
#pragma unroll
    for (int f = 0; f < 8; ++f) {
      const int mt = f >> 2, nt = f & 3;
      const int r0 = w * 32 + mt * 16 + kh;
      const int c0 = nt * 16 + nn;
#pragma unroll
      for (int r = 0; r < 8; ++r)
        mb[(r0 + r) * MB_ST + c0] = (__bf16)Mreg[f][r];
    }
    __syncthreads();

    // ---- phase B: pred = kc @ M[:,J], retr = qc @ M[:,J], K = 512 ----
    v8f acc0 = v8zero(), acc1 = v8zero();
    {
      const int srow = tid & 255;
      const int row = srow >> 2, seg = (srow & 3) * 8;
      const __bf16* asrc = ((tid < 256) ? kb : qb) +
                           (size_t)(b * Ssz + crow + row) * Dsz + seg;
      // dynamic-LDS base is offset 0 (no static __shared__ in this kernel)
      const unsigned ldst = (unsigned)((tid < 256) ? SM_AK : SM_AQ) +
                            (unsigned)(row * 80 + seg * 2);
      const __bf16* atile = (__bf16*)(smraw + (isPred ? SM_AK : SM_AQ)) +
                            strip * 40;
      for (int kk = 0; kk < Dsz; kk += 32) {
        // async global -> LDS copy of the 64x32 kc/qc tile (ASYNCcnt tracked)
        {
          unsigned long long ga = (unsigned long long)(size_t)(asrc + kk);
          asm volatile("global_load_async_to_lds_b128 %0, %1, off"
                       :: "v"(ldst), "v"(ga) : "memory");
        }
        if (kk + 32 < Dsz) __builtin_prefetch(asrc + kk + 32, 0, 0);
        asm volatile("s_wait_asynccnt 0x0" ::: "memory");
        __syncthreads();
        v16bf af = ld_frag_a(atile, 40, lane);
        v16bf b0 = ld_frag_b(mb + (size_t)kk * MB_ST + nhalf * 32, MB_ST, lane);
        v16bf b1 = ld_frag_b(mb + (size_t)kk * MB_ST + nhalf * 32 + 16, MB_ST, lane);
        acc0 = wmma_bf16(af, b0, acc0);
        acc1 = wmma_bf16(af, b1, acc1);
        __syncthreads();
      }
    }

    // ---- phase B epilogue: grad' = -LR*2*(pred-v)/D -> LDS ; retr -> out ----
    {
      const int c0 = strip + kh;
      const int ncol = nhalf * 32;
      if (isPred) {
#pragma unroll
        for (int ntl = 0; ntl < 2; ++ntl) {
          v8f a = ntl ? acc1 : acc0;
          const int n = ncol + ntl * 16 + nn;
#pragma unroll
          for (int r = 0; r < 8; ++r) {
            float vv = (float)vb[(size_t)(b * Ssz + crow + c0 + r) * Dsz + J + n];
            float g = (a[r] - vv) * (-0.000390625f);  // -LR * 2/D
            gr[(c0 + r) * MB_ST + n] = (__bf16)g;
          }
        }
      } else {
#pragma unroll
        for (int ntl = 0; ntl < 2; ++ntl) {
          v8f a = ntl ? acc1 : acc0;
          const int n = ncol + ntl * 16 + nn;
#pragma unroll
          for (int r = 0; r < 8; ++r)
            out[(size_t)(b * Ssz + crow + c0 + r) * Dsz + J + n] = a[r];
        }
      }
    }

    // ---- phase D: stage k_c^T into mb region via per-thread 8x8 register
    //      transpose (coalesced b128 loads, b128 LDS stores) ----
    {
      const int c0 = (tid >> 6) * 8;   // chunk rows (transposed cols)
      const int d0 = (tid & 63) * 8;   // feature rows of k^T
      union Row { uint4 q; __bf16 h[8]; };
      Row in[8];
#pragma unroll
      for (int i = 0; i < 8; ++i)
        in[i].q = *(const uint4*)(kb + (size_t)(b * Ssz + crow + c0 + i) * Dsz + d0);
#pragma unroll
      for (int j = 0; j < 8; ++j) {
        Row o;
#pragma unroll
        for (int i = 0; i < 8; ++i) o.h[i] = in[i].h[j];
        *(uint4*)&mb[(d0 + j) * MB_ST + c0] = o.q;
      }
    }
    __syncthreads();

    // ---- phase E: Smom' = 0.9*Smom + k^T @ grad' (momentum folded into C) ---
    v8f tmp[8];
#pragma unroll
    for (int f = 0; f < 8; ++f) tmp[f] = Sreg[f] * 0.9f;
    const int rowb = w * 32;
#pragma unroll
    for (int cki = 0; cki < 2; ++cki) {
      const int ck = cki * 32;
      v16bf A0 = ld_frag_a(mb + (size_t)rowb * MB_ST + ck, MB_ST, lane);
      v16bf A1 = ld_frag_a(mb + (size_t)(rowb + 16) * MB_ST + ck, MB_ST, lane);
#pragma unroll
      for (int nt = 0; nt < 4; ++nt) {
        v16bf Bf = ld_frag_b(gr + (size_t)ck * MB_ST + nt * 16, MB_ST, lane);
        tmp[nt]     = wmma_bf16(A0, Bf, tmp[nt]);
        tmp[4 + nt] = wmma_bf16(A1, Bf, tmp[4 + nt]);
      }
    }

    // ---- phase F: register update of M/Smom + surprise (||dW||^2 partial) ---
    float ssum = 0.0f;
#pragma unroll
    for (int f = 0; f < 8; ++f) {
#pragma unroll
      for (int r = 0; r < 8; ++r) {
        float dlt = tmp[f][r] - 0.9f * Sreg[f][r];  // = -LR * dW element
        ssum += dlt * dlt;
        Mreg[f][r] = 0.99f * Mreg[f][r] + tmp[f][r];
      }
      Sreg[f] = tmp[f];
    }
    ssum *= 100.0f;  // (1/LR)^2
#pragma unroll
    for (int off = 16; off > 0; off >>= 1) ssum += __shfl_xor(ssum, off);
    if (lane == 0) red[w] = ssum;
    __syncthreads();
    if (tid == 0) {
      float tot = 0.0f;
#pragma unroll
      for (int i = 0; i < 16; ++i) tot += red[i];
      atomicAdd(&surp[b * NCH + t], tot);
    }
  }
}

__global__ void finish_kernel(const float* __restrict__ s, float* __restrict__ o) {
  int i = threadIdx.x;
  if (i < Bsz * NCH) o[i] = sqrtf(s[i]);
}

extern "C" void kernel_launch(void* const* d_in, const int* in_sizes, int n_in,
                              void* d_out, int out_size, void* d_ws, size_t ws_size,
                              hipStream_t stream) {
  const float* x  = (const float*)d_in[0];
  const float* Wk = (const float*)d_in[1];
  const float* Wv = (const float*)d_in[2];
  const float* Wq = (const float*)d_in[3];

  const size_t NB = (size_t)Bsz * Ssz * Dsz;
  __bf16* kb = (__bf16*)d_ws;
  __bf16* vb = kb + NB;
  __bf16* qb = vb + NB;
  float* surp = (float*)(qb + NB);

  hipMemsetAsync(surp, 0, Bsz * NCH * sizeof(float), stream);

  proj_kernel<<<dim3((Bsz * Ssz) / 128, Dsz / 64, 3), 256, 0, stream>>>(
      x, Wk, Wv, Wq, kb, vb, qb);

  (void)hipFuncSetAttribute((const void*)step_kernel,
                            hipFuncAttributeMaxDynamicSharedMemorySize,
                            SMEM_BYTES);
  step_kernel<<<dim3(Dsz / 64, Bsz), 512, SMEM_BYTES, stream>>>(
      kb, vb, qb, (float*)d_out, surp);

  finish_kernel<<<1, 256, 0, stream>>>(surp, (float*)d_out + NB);
}
